// MultiScale_GraphConv_17523466567845
// MI455X (gfx1250) — compile-verified, matrix operand order checked by reading, not run on hardware
//
#include <hip/hip_runtime.h>
#include <hip/hip_bf16.h>

// ---------------------------------------------------------------------------
// MultiScale GraphConv on gfx1250 (MI455X)
//   Stage A (prep):  xT = bf16(x)^T        [C][N]
//                    WcT = bf16( (Wp[j] @ Wm_j)^T )   [OUT][3*C]
//                    cst[o] = bm[o] + sum_k bp[k]*Wm[k,o]
//   Stage B (agg):   per (b,j,16-row tile): scatter edges into LDS Weff tile
//                    (ds_add_f32), then WMMA bf16  [16x2048]@[2048x128]
//                    -> cum[b,f,j,c]  (stored pre-permuted for stage C)
//   Stage C (mlp):   out = cum[8192x384] @ WcT^T (bf16 WMMA) + cst
// ---------------------------------------------------------------------------

typedef __attribute__((ext_vector_type(16))) __bf16 v16bf;
typedef __attribute__((ext_vector_type(8)))  float  v8f;
typedef __attribute__((ext_vector_type(4)))  float  f4;
typedef __attribute__((ext_vector_type(4)))  int    i4;

#define Bn 4
#define Sn 4
#define Nn 2048
#define Cn 128
#define On 256
#define Jn 3          // S-1
#define Kt 384        // Jn*Cn

// ---------------- prep kernels (tiny) ----------------

__global__ void k_prep_xT(const float* __restrict__ x, __bf16* __restrict__ xT) {
    int g = blockIdx.x * blockDim.x + threadIdx.x;     // 0 .. N*C-1
    int t = g / Cn, c = g % Cn;
    xT[c * Nn + t] = (__bf16)x[g];
}

__global__ void k_prep_wc(const float* __restrict__ Wp, const float* __restrict__ Wm,
                          __bf16* __restrict__ WcT) {
    int g = blockIdx.x * blockDim.x + threadIdx.x;     // 0 .. Kt*On-1
    int k = g / On, o = g % On;
    int j = k / Cn, c = k % Cn;
    const float* wp = Wp + (j * Cn + c) * Cn;          // row over d (contiguous)
    const float* wm = Wm + (size_t)j * Cn * On + o;    // column over d (stride On)
    float s = 0.f;
    #pragma unroll 4
    for (int d = 0; d < Cn; ++d) s += wp[d] * wm[(size_t)d * On];
    WcT[(size_t)o * Kt + k] = (__bf16)s;
}

__global__ void k_prep_const(const float* __restrict__ bp, const float* __restrict__ Wm,
                             const float* __restrict__ bm, float* __restrict__ cst) {
    int o = threadIdx.x;                               // 256 threads
    float s = bm[o];
    for (int k = 0; k < Kt; ++k) s += bp[k] * Wm[(size_t)k * On + o];
    cst[o] = s;
}

// ---------------- stage B: scatter + WMMA aggregation ----------------
// grid = B*Jn*(N/16) blocks of 128 threads; dynamic LDS = 16*N floats (128 KB)

__global__ void k_aggregate(const float* __restrict__ A_powers,
                            const int*   __restrict__ A_last_edge,
                            const __bf16* __restrict__ xT,
                            float* __restrict__ cum /* [B][N][Jn][Cn] */) {
    extern __shared__ float w_lds[];                   // Weff tile [16][Nn]
    const int blk   = blockIdx.x;
    const int ftile = blk & 127;
    const int bj    = blk >> 7;
    const int j     = bj % Jn;
    const int b     = bj / Jn;
    const int fbase = ftile * 16;
    const int tid   = threadIdx.x;

    // ---- zero the Weff tile ----
    f4* wl4 = (f4*)w_lds;
    f4 zero4 = {};
    for (int i = tid; i < 16 * Nn / 4; i += 128) wl4[i] = zero4;
    __syncthreads();

    // ---- scatter: Weff[m][edge-1] += A_powers[b,j+1,f,to]  (LDS atomics) ----
    const float* pw = A_powers   + (((size_t)b * Sn + (j + 1)) * Nn + fbase) * Nn;
    const int*   pe = A_last_edge + (((size_t)b * Jn + j) * Nn + fbase) * Nn;
    for (int m = 0; m < 16; ++m) {
        const f4* prow_w = (const f4*)(pw + (size_t)m * Nn);
        const i4* prow_e = (const i4*)(pe + (size_t)m * Nn);
        float* wrow = w_lds + m * Nn;
        for (int base = tid; base < Nn / 4; base += 128) {
            f4 wv = __builtin_nontemporal_load(prow_w + base);  // TH=NT: one-shot stream
            i4 ev = __builtin_nontemporal_load(prow_e + base);
            if (ev.x) atomicAdd(&wrow[ev.x - 1], wv.x);
            if (ev.y) atomicAdd(&wrow[ev.y - 1], wv.y);
            if (ev.z) atomicAdd(&wrow[ev.z - 1], wv.z);
            if (ev.w) atomicAdd(&wrow[ev.w - 1], wv.w);
        }
    }
    __syncthreads();

    // ---- WMMA: [16 x 2048](bf16 from LDS) @ x[2048 x 128] -> [16 x 128] ----
    const int lane = tid & 31;
    const int wave = tid >> 5;          // 4 waves, each owns 2 n-tiles (32 cols of C)
    const int half = lane >> 4;         // K sub-range select (contiguous k = 16*half + s)
    const int l16  = lane & 15;
    const int n0 = (wave * 2 + 0) * 16 + l16;
    const int n1 = (wave * 2 + 1) * 16 + l16;
    v8f acc0 = {}; v8f acc1 = {};

    for (int kb = 0; kb < Nn; kb += 32) {
        const int ks = kb + half * 16;
        // A fragment: lane holds M=l16, K=ks..ks+15 (f32 -> bf16)
        v16bf a;
        const f4* ap = (const f4*)(w_lds + l16 * Nn + ks);
        #pragma unroll
        for (int q = 0; q < 4; ++q) {
            f4 v = ap[q];
            a[q * 4 + 0] = (__bf16)v.x; a[q * 4 + 1] = (__bf16)v.y;
            a[q * 4 + 2] = (__bf16)v.z; a[q * 4 + 3] = (__bf16)v.w;
        }
        // B fragments: lane holds N=n, K=ks..ks+15, contiguous in xT (L2-hot)
        v16bf b0 = *(const v16bf*)(xT + (size_t)n0 * Nn + ks);
        v16bf b1 = *(const v16bf*)(xT + (size_t)n1 * Nn + ks);
        acc0 = __builtin_amdgcn_wmma_f32_16x16x32_bf16(false, a, false, b0,
                                                       (short)0, acc0, false, false);
        acc1 = __builtin_amdgcn_wmma_f32_16x16x32_bf16(false, a, false, b1,
                                                       (short)0, acc1, false, false);
    }

    // ---- epilogue: store pre-permuted as cum[b, f, j, c] ----
    #pragma unroll
    for (int r = 0; r < 8; ++r) {
        const int mo = r + 8 * half;                       // C/D layout
        const size_t row = (size_t)b * Nn + fbase + mo;
        cum[(row * Jn + j) * Cn + (wave * 2 + 0) * 16 + l16] = acc0[r];
        cum[(row * Jn + j) * Cn + (wave * 2 + 1) * 16 + l16] = acc1[r];
    }
}

// ---------------- stage C: fused (per-scale Linear + MLP) GEMM ----------------
// out[8192 x 256] = cum[8192 x 384] @ Wc[384 x 256] + cst ; grid = 8192/16 blocks

__global__ void k_gemm2(const float*  __restrict__ cum,   // [B*N][Kt]
                        const __bf16* __restrict__ WcT,   // [On][Kt]
                        const float*  __restrict__ cst,   // [On]
                        float* __restrict__ out) {        // [B*N][On]
    const int mbase = blockIdx.x * 16;
    const int tid  = threadIdx.x;
    const int lane = tid & 31;
    const int wave = tid >> 5;          // 4 waves x 4 n-tiles = 256 output cols
    const int half = lane >> 4;
    const int l16  = lane & 15;
    v8f acc[4] = {{}, {}, {}, {}};

    for (int kb = 0; kb < Kt; kb += 32) {
        const int ks = kb + half * 16;
        v16bf a;
        const f4* ap = (const f4*)(cum + (size_t)(mbase + l16) * Kt + ks);
        #pragma unroll
        for (int q = 0; q < 4; ++q) {
            f4 v = ap[q];
            a[q * 4 + 0] = (__bf16)v.x; a[q * 4 + 1] = (__bf16)v.y;
            a[q * 4 + 2] = (__bf16)v.z; a[q * 4 + 3] = (__bf16)v.w;
        }
        #pragma unroll
        for (int t = 0; t < 4; ++t) {
            const int n = (wave * 4 + t) * 16 + l16;
            v16bf bm = *(const v16bf*)(WcT + (size_t)n * Kt + ks);
            acc[t] = __builtin_amdgcn_wmma_f32_16x16x32_bf16(false, a, false, bm,
                                                             (short)0, acc[t], false, false);
        }
    }

    #pragma unroll
    for (int t = 0; t < 4; ++t) {
        const int nb = (wave * 4 + t) * 16;
        const float cv = cst[nb + l16];
        #pragma unroll
        for (int r = 0; r < 8; ++r) {
            const int mo = r + 8 * half;
            out[(size_t)(mbase + mo) * On + nb + l16] = acc[t][r] + cv;
        }
    }
}

// ---------------- host launcher ----------------

extern "C" void kernel_launch(void* const* d_in, const int* in_sizes, int n_in,
                              void* d_out, int out_size, void* d_ws, size_t ws_size,
                              hipStream_t stream) {
    // setup_inputs order:
    // 0:A_binary 1:A_powers 2:A_lookup 3:A_last_edge 4:x 5:Wp 6:bp 7:Wm 8:bm
    const float* A_powers = (const float*)d_in[1];
    const int*   A_last   = (const int*)  d_in[3];
    const float* x        = (const float*)d_in[4];
    const float* Wp       = (const float*)d_in[5];
    const float* bp       = (const float*)d_in[6];
    const float* Wm       = (const float*)d_in[7];
    const float* bm       = (const float*)d_in[8];
    float* out = (float*)d_out;

    char* ws = (char*)d_ws;
    __bf16* xT  = (__bf16*)(ws);                              // 512 KB
    __bf16* WcT = (__bf16*)(ws + (512 << 10));                // 192 KB
    float*  cst = (float*) (ws + (512 << 10) + (192 << 10));  // 1 KB
    float*  cum = (float*) (ws + (1 << 20));                  // 12 MB, [B*N][Kt]

    k_prep_xT   <<<(Nn * Cn) / 256, 256, 0, stream>>>(x, xT);
    k_prep_wc   <<<(Kt * On) / 128, 128, 0, stream>>>(Wp, Wm, WcT);
    k_prep_const<<<1, On, 0, stream>>>(bp, Wm, bm, cst);

    k_aggregate <<<Bn * Jn * (Nn / 16), 128, 16 * Nn * sizeof(float), stream>>>(
        A_powers, A_last, xT, cum);

    k_gemm2     <<<(Bn * Nn) / 16, 128, 0, stream>>>(cum, WcT, cst, out);
}
